// GAT_EFA_15195594293708
// MI455X (gfx1250) — matrix operand order repaired and checked by reading, not compiled, not run on hardware
//
#include <hip/hip_runtime.h>
#include <hip/hip_bf16.h>

// ---------------------------------------------------------------------------
// GAT forward for MI455X (gfx1250, CDNA5, wave32, WMMA).
// Heavy GEMMs use v_wmma_f32_16x16x32_f16 (f16 in, f32 accumulate).
// Edge tensor kept in f16 (82MB) so it lives in the 192MB L2 across all
// 12 attention head passes.  f=[...,64] per-edge activations are never
// materialized: cp is reduced out of WMMA accumulators in-register.
// B operands are pre-swizzled into fragment-contiguous layout so every
// fragment load is two global_load_b128 (no scalar u16 gathers).
// ---------------------------------------------------------------------------

#define ALPHA 0.2f
#define NEGV  -9e15f

#define BSZ   8
#define NN    200
#define BN    1600            // BSZ*NN
#define NEDGE 320000          // BSZ*NN*NN

typedef __attribute__((ext_vector_type(16))) _Float16 v16h;
typedef __attribute__((ext_vector_type(8)))  _Float16 v8h;
typedef __attribute__((ext_vector_type(8)))  float    v8f;

__device__ __forceinline__ float reluf(float x){ return x > 0.f ? x : 0.f; }
__device__ __forceinline__ float leakyf(float x){ return x > 0.f ? x : ALPHA * x; }

// ---- WMMA fragment loaders (layouts per CDNA5 ISA 7.12.2) ------------------
// A (16x32 f16): lane L holds row m=L%16; with hi=L/16, elements 0..7 are
// K = kbase+hi*8+e, elements 8..15 are K = kbase+16+hi*8+e.
// Two 16-byte vector loads -> global_load_b128.
__device__ __forceinline__ v16h load_a_frag(const _Float16* __restrict__ A,
                                            long long row, int ldk, int kbase,
                                            bool valid, int lane) {
  int hi = (lane >> 4) & 1;
  if (valid) {
    const _Float16* p = A + row * (long long)ldk + kbase + hi * 8;
    v8h lo = *(const v8h*)(p);
    v8h hh = *(const v8h*)(p + 16);
    return __builtin_shufflevector(lo, hh, 0,1,2,3,4,5,6,7,8,9,10,11,12,13,14,15);
  }
  v16h z = {};
  return z;
}

// B packed layout: for fragment f (= kbIdx*nnt + ntg), lane L owns 16
// contiguous f16 at Bp[(f*32 + L)*16 + e], where element e corresponds to
// B[k = kbIdx*32 + (L/16)*8 + e for e<8, else kbIdx*32 + 16 + (L/16)*8 + e-8, n = ntg*16 + L%16].
__device__ __forceinline__ v16h load_b_packed(const _Float16* __restrict__ Bp,
                                              int frag, int lane) {
  const _Float16* p = Bp + ((long long)frag * 32 + lane) * 16;
  v8h lo = *(const v8h*)(p);
  v8h hh = *(const v8h*)(p + 8);
  return __builtin_shufflevector(lo, hh, 0,1,2,3,4,5,6,7,8,9,10,11,12,13,14,15);
}

// ---- small utility kernels -------------------------------------------------
__global__ void k_f32_to_f16(const float* __restrict__ in, _Float16* __restrict__ out, long long n) {
  long long i = (long long)blockIdx.x * 256 + threadIdx.x;
  if (i < n) out[i] = (_Float16)in[i];
}

// pack f32 row-major B [z][K][N] into f16 fragment-contiguous layout
__global__ void k_pack_b(const float* __restrict__ W, _Float16* __restrict__ out,
                         int K, int N, long long total, long long perZ) {
  long long i = (long long)blockIdx.x * 256 + threadIdx.x;
  if (i >= total) return;
  long long z = i / perZ;
  long long r = i % perZ;
  int e    = (int)(r & 15);
  long long t = r >> 4;
  int lane = (int)(t & 31);
  long long f = t >> 5;
  int nnt = N / 16;
  int kb  = (int)(f / nnt);
  int ntg = (int)(f % nnt);
  int k = kb * 32 + ((e < 8) ? ((lane >> 4) * 8 + e) : (16 + (lane >> 4) * 8 + (e - 8)));
  int n = ntg * 16 + (lane & 15);
  out[i] = (_Float16)W[z * perZ + (long long)k * N + n];
}

// colsum over the 10 tiled rows: out[c] = sum_{f<10} w[f*C + c]
__global__ void k_colsum10(const float* __restrict__ w, float* __restrict__ out, int C) {
  int c = blockIdx.x * 256 + threadIdx.x;
  if (c < C) { float s = 0.f; for (int f = 0; f < 10; ++f) s += w[f * C + c]; out[c] = s; }
}

// ---- node features: x = concat(xl, xd, xt, xu)  [bs,N,128] (f32 + f16) -----
__global__ void k_node_features(const float* __restrict__ xloc, const float* __restrict__ xtw,
                                const float* __restrict__ xd,   const float* __restrict__ xdual,
                                const float* __restrict__ w1, const float* __restrict__ b1,
                                const float* __restrict__ w2, const float* __restrict__ b2,
                                const float* __restrict__ w3, const float* __restrict__ b3,
                                const float* __restrict__ w4, const float* __restrict__ b4,
                                float* __restrict__ xcat, _Float16* __restrict__ xcat16) {
  int bn = blockIdx.x;          // 0..1599
  int c  = threadIdx.x;         // 0..127
  int seg = c >> 5, cc = c & 31;
  float acc;
  if (seg == 0) {               // xl: tile(x_loc,FG) @ node_w1 [20,32]
    acc = b1[cc];
    for (int f = 0; f < 20; ++f) acc += xloc[bn * 2 + (f & 1)] * w1[f * 32 + cc];
  } else if (seg == 1) {        // xd: tile(x_d,FG) @ node_w2 [10,32]
    acc = b2[cc];
    for (int f = 0; f < 10; ++f) acc += xd[bn] * w2[f * 32 + cc];
  } else if (seg == 2) {        // xt
    acc = b3[cc];
    for (int f = 0; f < 20; ++f) acc += xtw[bn * 2 + (f & 1)] * w3[f * 32 + cc];
  } else {                      // xu
    acc = b4[cc];
    for (int f = 0; f < 10; ++f) acc += xdual[bn] * w4[f * 32 + cc];
  }
  acc = reluf(acc);
  xcat[(long long)bn * 128 + c]   = acc;
  xcat16[(long long)bn * 128 + c] = (_Float16)acc;
}

// ---- generic WMMA GEMM: C = act(A[MxK] @ Bp + bias), z-batched over B ------
__global__ __launch_bounds__(32) void k_gemm(const _Float16* __restrict__ A,
                                             const _Float16* __restrict__ Bp,
                                             const float* __restrict__ bias,
                                             float* __restrict__ Cf32,
                                             _Float16* __restrict__ Cf16,
                                             int M, int K, int N, int act,
                                             long long bStrideZ, long long biasStrideZ,
                                             long long cStrideZ) {
  int lane = threadIdx.x;
  int row0 = blockIdx.x * 16;
  int n0   = blockIdx.y * 64;
  int z    = blockIdx.z;
  const _Float16* Bz = Bp + bStrideZ * z;
  const float* biasZ = bias ? bias + biasStrideZ * z : nullptr;
  long long coff = cStrideZ * z;
  int hi = lane >> 4, nl = lane & 15;
  int nnt = N / 16;

  v8f acc[4] = {};
  long long rowA = row0 + nl;
  bool valid = rowA < M;
  for (int kb = 0; kb < K; kb += 32) {
    v16h a = load_a_frag(A, rowA, K, kb, valid, lane);
    int fbase = (kb >> 5) * nnt + (n0 >> 4);
#pragma unroll
    for (int nt = 0; nt < 4; ++nt) {
      v16h b = load_b_packed(Bz, fbase + nt, lane);
      acc[nt] = __builtin_amdgcn_wmma_f32_16x16x32_f16(false, a, false, b, (short)0, acc[nt], false, false);
    }
  }
#pragma unroll
  for (int nt = 0; nt < 4; ++nt) {
    int n = n0 + nt * 16 + nl;
    float bv = biasZ ? biasZ[n] : 0.f;
#pragma unroll
    for (int r = 0; r < 8; ++r) {
      long long m = row0 + r + 8 * hi;
      if (m < M) {
        float v = acc[nt][r] + bv;
        if (act == 1) v = reluf(v); else if (act == 2) v = leakyf(v);
        long long idx = coff + m * (long long)N + n;
        if (Cf32) Cf32[idx] = v;
        if (Cf16) Cf16[idx] = (_Float16)v;
      }
    }
  }
}

// ---- fused edge features + edge_emb ----------------------------------------
// ef_raw[e,k] = k<64 ? relu(d*S1[k]+b1[k]) : relu(a*S2[k-64]+b2[k-64])
// ef16 = relu(ef_raw @ W[128,128] + bias) stored f16.
__global__ __launch_bounds__(32) void k_edge_emb(const float* __restrict__ dis,
                                                 const float* __restrict__ adj,
                                                 const float* __restrict__ S1, const float* __restrict__ b1,
                                                 const float* __restrict__ S2, const float* __restrict__ b2,
                                                 const _Float16* __restrict__ Wp,  // packed, K=128 N=128
                                                 const float* __restrict__ bias,
                                                 _Float16* __restrict__ ef16) {
  int lane = threadIdx.x;
  int hi = lane >> 4, nl = lane & 15;
  long long e = (long long)blockIdx.x * 16 + nl;
  int n0 = blockIdx.y * 64;
  float d = dis[e], a = adj[e];
  v8f acc[4] = {};
#pragma unroll
  for (int kb = 0; kb < 128; kb += 32) {
    v16h afr;
#pragma unroll
    for (int e2 = 0; e2 < 16; ++e2) {
      int k = kb + ((e2 < 8) ? (hi * 8 + e2) : (16 + hi * 8 + (e2 - 8)));
      float v = (k < 64) ? reluf(d * S1[k] + b1[k]) : reluf(a * S2[k - 64] + b2[k - 64]);
      afr[e2] = (_Float16)v;
    }
    int fbase = (kb >> 5) * 8 + (n0 >> 4);
#pragma unroll
    for (int nt = 0; nt < 4; ++nt) {
      v16h bfr = load_b_packed(Wp, fbase + nt, lane);
      acc[nt] = __builtin_amdgcn_wmma_f32_16x16x32_f16(false, afr, false, bfr, (short)0, acc[nt], false, false);
    }
  }
#pragma unroll
  for (int nt = 0; nt < 4; ++nt) {
    int n = n0 + nt * 16 + nl;
    float bv = bias[n];
#pragma unroll
    for (int r = 0; r < 8; ++r) {
      long long row = (long long)blockIdx.x * 16 + r + 8 * hi;
      ef16[row * 128 + n] = (_Float16)reluf(acc[nt][r] + bv);
    }
  }
}

// ---- collapse linear attention terms: v1 = W1@a, c1 = b1·a -----------------
__global__ void k_sprep(const float* __restrict__ w1, const float* __restrict__ b1,
                        const float* __restrict__ w2, const float* __restrict__ b2,
                        const float* __restrict__ ahw,
                        float* __restrict__ v1, float* __restrict__ v2,
                        float* __restrict__ c1, float* __restrict__ c2, int din) {
  int hd = blockIdx.x; int k = threadIdx.x;
  const float* a = ahw + hd * 64;
  if (k < din) {
    const float* W1 = w1 + ((long long)hd * din + k) * 64;
    const float* W2 = w2 + ((long long)hd * din + k) * 64;
    float s1 = 0.f, s2 = 0.f;
    for (int c = 0; c < 64; ++c) { s1 += W1[c] * a[c]; s2 += W2[c] * a[c]; }
    v1[hd * din + k] = s1; v2[hd * din + k] = s2;
  }
  if (k == 0) {
    float t1 = 0.f, t2 = 0.f;
    for (int c = 0; c < 64; ++c) { t1 += b1[hd * 64 + c] * a[c]; t2 += b2[hd * 64 + c] * a[c]; }
    c1[hd] = t1; c2[hd] = t2;
  }
}

// s1[hd,bn] = x[bn,:] . v1[hd,:] + c1[hd]  (same for s2)
__global__ __launch_bounds__(32) void k_scompute(const float* __restrict__ x,
                                                 const float* __restrict__ v1, const float* __restrict__ v2,
                                                 const float* __restrict__ c1, const float* __restrict__ c2,
                                                 float* __restrict__ s1, float* __restrict__ s2, int din) {
  int bn = blockIdx.x, hd = blockIdx.y, lane = threadIdx.x;
  const float* xr = x + (long long)bn * din;
  const float* V1 = v1 + hd * din;
  const float* V2 = v2 + hd * din;
  float a = 0.f, b = 0.f;
  for (int k = lane; k < din; k += 32) { float xv = xr[k]; a += xv * V1[k]; b += xv * V2[k]; }
  for (int m = 16; m; m >>= 1) { a += __shfl_xor(a, m); b += __shfl_xor(b, m); }
  if (lane == 0) { s1[hd * BN + bn] = a + c1[hd]; s2[hd * BN + bn] = b + c2[hd]; }
}

// ---- fused attention head: f (WMMA) -> cp -> softmax -> att@h --------------
// one wave per (row i, batch b, head hd)
__global__ __launch_bounds__(32) void k_attention(const _Float16* __restrict__ ef16,
                                                  const _Float16* __restrict__ wf16p, // packed [H][128x64]
                                                  const float* __restrict__ wf_b,     // [H][64]
                                                  const float* __restrict__ bfp,      // [H][64]
                                                  const float* __restrict__ af_w,     // [H][64]
                                                  const float* __restrict__ af_b,     // [H]
                                                  const float* __restrict__ ah_b,     // [H]
                                                  const float* __restrict__ s1,       // [H][BN]
                                                  const float* __restrict__ s2,       // [H][BN]
                                                  const float* __restrict__ hmat,     // [H][BN][64]
                                                  const float* __restrict__ adj,      // [NEDGE]
                                                  float* __restrict__ hp)             // [H][BN][64]
{
  int i = blockIdx.x, b = blockIdx.y, hd = blockIdx.z;
  int lane = threadIdx.x;
  int hi = lane >> 4, nl = lane & 15;
  __shared__ float score[208];

  // preload all 16 B fragments (weights stay in VGPRs across the j loop)
  const _Float16* Wp = wf16p + (long long)hd * 128 * 64;
  v16h bfrag[4][4];
#pragma unroll
  for (int ks = 0; ks < 4; ++ks)
#pragma unroll
    for (int nt = 0; nt < 4; ++nt)
      bfrag[ks][nt] = load_b_packed(Wp, ks * 4 + nt, lane);

  float fb[4], afv[4];
#pragma unroll
  for (int nt = 0; nt < 4; ++nt) {
    int n = nt * 16 + nl;
    fb[nt]  = wf_b[hd * 64 + n] + bfp[hd * 64 + n];
    afv[nt] = af_w[hd * 64 + n];
  }
  float afb = af_b[hd], ahb = ah_b[hd];
  long long bn0 = (long long)b * NN;
  float s1i = s1[(long long)hd * BN + bn0 + i];
  long long erow0 = (bn0 + i) * NN;

  for (int jt = 0; jt < 13; ++jt) {
    int j0 = jt * 16;
    // prefetch next tile's edge rows into cache (global_prefetch_b8)
    if (j0 + 16 + nl < NN)
      __builtin_prefetch(ef16 + (erow0 + j0 + 16 + nl) * 128, 0, 0);
    v8f acc[4] = {};
    long long rowe = erow0 + j0 + nl;
    bool valid = (j0 + nl) < NN;
#pragma unroll
    for (int ks = 0; ks < 4; ++ks) {
      v16h a = load_a_frag(ef16, rowe, 128, ks * 32, valid, lane);
#pragma unroll
      for (int nt = 0; nt < 4; ++nt)
        acc[nt] = __builtin_amdgcn_wmma_f32_16x16x32_f16(false, a, false, bfrag[ks][nt], (short)0, acc[nt], false, false);
    }
    // cp[m] = leaky( sum_n leaky(f[m,n]) * af[n] + af_b ); reduce N over 16 lanes
    float cp[8];
#pragma unroll
    for (int r = 0; r < 8; ++r) {
      float t = 0.f;
#pragma unroll
      for (int nt = 0; nt < 4; ++nt) t += leakyf(acc[nt][r] + fb[nt]) * afv[nt];
      t += __shfl_xor(t, 1); t += __shfl_xor(t, 2); t += __shfl_xor(t, 4); t += __shfl_xor(t, 8);
      cp[r] = leakyf(t + afb);
    }
    if (nl == 0) {
#pragma unroll
      for (int r = 0; r < 8; ++r) {
        int j = j0 + r + 8 * hi;
        float sc = NEGV;
        if (j < NN) {
          float c = leakyf(s1i + s2[(long long)hd * BN + bn0 + j] + ahb);
          sc = (adj[erow0 + j] > 0.f) ? (c + cp[r]) : NEGV;
        }
        if (j < 208) score[j] = sc;
      }
    }
  }
  __syncthreads();
  // softmax over j<200
  float mx = -3.4e38f;
  for (int j = lane; j < NN; j += 32) mx = fmaxf(mx, score[j]);
  for (int m = 16; m; m >>= 1) mx = fmaxf(mx, __shfl_xor(mx, m));
  float sum = 0.f;
  for (int j = lane; j < NN; j += 32) sum += __expf(score[j] - mx);
  for (int m = 16; m; m >>= 1) sum += __shfl_xor(sum, m);
  float inv = 1.f / sum;
  for (int j = lane; j < NN; j += 32) score[j] = __expf(score[j] - mx) * inv;
  __syncthreads();
  // hp[i,:] = att @ h : lane covers 2 columns (float2 -> global_load_b64)
  const float2* h2 = (const float2*)(hmat + ((long long)hd * BN + bn0) * 64);
  float2 acc2; acc2.x = 0.f; acc2.y = 0.f;
  for (int j = 0; j < NN; ++j) {
    float av = score[j];
    float2 hv = h2[(long long)j * 32 + lane];
    acc2.x += av * hv.x;
    acc2.y += av * hv.y;
  }
  float2* o2 = (float2*)(hp + ((long long)hd * BN + bn0 + i) * 64);
  o2[lane] = acc2;
}

// ---- per-node normalization over (batch, feat) -----------------------------
// mode 0: out[b,i,hd*64+c] = leaky(norm)   (layer0 concat)
// mode 1: in-place normalize hp            (layer1)
__global__ void k_norm(float* __restrict__ hp, float* __restrict__ xout, int H, int mode) {
  int i = blockIdx.x, hd = blockIdx.y, t = threadIdx.x;   // 256 threads
  __shared__ float ssum[256], ssq[256];
  float s = 0.f, q = 0.f, vals[2];
#pragma unroll
  for (int u = 0; u < 2; ++u) {
    int v = t + u * 256; int b = v >> 6; int c = v & 63;
    float x = hp[(((long long)hd * BSZ + b) * NN + i) * 64 + c];
    vals[u] = x; s += x; q += x * x;
  }
  ssum[t] = s; ssq[t] = q; __syncthreads();
  for (int st = 128; st; st >>= 1) {
    if (t < st) { ssum[t] += ssum[t + st]; ssq[t] += ssq[t + st]; }
    __syncthreads();
  }
  float m = ssum[0] / 512.f;
  float var = ssq[0] / 512.f - m * m;
  float invs = rsqrtf(1e-5f + var);
#pragma unroll
  for (int u = 0; u < 2; ++u) {
    int v = t + u * 256; int b = v >> 6; int c = v & 63;
    float nv = (vals[u] - m) * invs;
    if (mode == 0) xout[((long long)b * NN + i) * (H * 64) + hd * 64 + c] = leakyf(nv);
    else           hp[(((long long)hd * BSZ + b) * NN + i) * 64 + c] = nv;
  }
}

// layer1: x1 = relu(mean over 4 heads)
__global__ void k_combine1(const float* __restrict__ hp1, float* __restrict__ x1) {
  int bn = blockIdx.x, c = threadIdx.x;
  float s = 0.f;
  for (int hd = 0; hd < 4; ++hd) s += hp1[((long long)hd * BN + bn) * 64 + c];
  x1[(long long)bn * 64 + c] = reluf(0.25f * s);
}

// ---- tail MLP + column features + softmax ----------------------------------
__global__ void k_tail(const float* __restrict__ x1,
                       const float* __restrict__ xw1w, const float* __restrict__ xw1b,
                       const float* __restrict__ xw2w, const float* __restrict__ xw2b,
                       const float* __restrict__ Src, const float* __restrict__ rcb,
                       const float* __restrict__ Sdi, const float* __restrict__ dib,
                       const float* __restrict__ pew, const float* __restrict__ peb,
                       const float* __restrict__ Spw, const float* __restrict__ pwb,
                       const float* __restrict__ colw, const float* __restrict__ colb,
                       const float* __restrict__ o1w, const float* __restrict__ o1b,
                       const float* __restrict__ o2w, const float* __restrict__ o2b,
                       const float* __restrict__ rcost, const float* __restrict__ dist,
                       const float* __restrict__ ponehot, float* __restrict__ out) {
  int b = blockIdx.x, t = threadIdx.x;  // 256 threads
  __shared__ float xa[200], feat[256], tmp[256], y1[128];
  for (int n = t; n < NN; n += 256) {
    float s = xw1b[0];
    const float* xr = x1 + ((long long)b * NN + n) * 64;
    for (int k = 0; k < 64; ++k) s += xr[k] * xw1w[k];
    xa[n] = reluf(s);
  }
  __syncthreads();
  if (t < 128) {
    float s = xw2b[t];
    for (int n = 0; n < NN; ++n) s += xa[n] * xw2w[n * 128 + t];
    feat[t] = reluf(s);
  }
  // path scalar (recomputed per thread; tiny)
  float xp;
  { float s = peb[0];
    for (int n = 0; n < NN; ++n) s += ponehot[b * NN + n] * pew[n];
    xp = reluf(s); }
  float rcv = rcost[b], div = dist[b];
  if (t < 64)       tmp[t] = reluf(rcv * Src[t] + rcb[t]);
  else if (t < 128) tmp[t] = reluf(div * Sdi[t - 64] + dib[t - 64]);
  else              tmp[t] = reluf(xp * Spw[t - 128] + pwb[t - 128]);
  __syncthreads();
  if (t < 128) {
    float s = colb[t];
    for (int q = 0; q < 256; ++q) s += tmp[q] * colw[q * 128 + t];
    feat[128 + t] = reluf(s);
  }
  __syncthreads();
  if (t < 128) {
    float s = o1b[t];
    for (int q = 0; q < 256; ++q) s += feat[q] * o1w[q * 128 + t];
    y1[t] = reluf(s);
  }
  __syncthreads();
  if (t == 0) {
    float z0 = o2b[0], z1 = o2b[1];
    for (int q = 0; q < 128; ++q) { z0 += y1[q] * o2w[q * 2]; z1 += y1[q] * o2w[q * 2 + 1]; }
    float mx = fmaxf(z0, z1);
    float e0 = __expf(z0 - mx), e1 = __expf(z1 - mx);
    float inv = 1.f / (e0 + e1);
    out[b * 2] = e0 * inv; out[b * 2 + 1] = e1 * inv;
  }
}

// ---------------------------------------------------------------------------
// Input index map (setup_inputs() dict-order flattening, depth-first):
enum {
  I_XLOC = 0, I_XTW, I_XD, I_DIS, I_ADJ, I_XDUAL, I_RC, I_DIST, I_PONE,
  I_NW1W, I_NW1B, I_NW2W, I_NW2B, I_NW3W, I_NW3B, I_NW4W, I_NW4B,
  I_EW1W, I_EW1B, I_EW2W, I_EW2B,
  I_NEMBW, I_NEMBB, I_EEMBW, I_EEMBB,
  I_RCWW, I_RCWB, I_DIWW, I_DIWB, I_PEW, I_PEB, I_PWW, I_PWB,
  I_COLW, I_COLB, I_XW1W, I_XW1B, I_XW2W, I_XW2B,
  I_O1W, I_O1B, I_O2W, I_O2B,
  // gat0 (H=8, din=128)
  G0_WHW, G0_WHB, G0_W1W, G0_W1B, G0_W2W, G0_W2B, G0_AHW, G0_AHB,
  G0_WFW, G0_WFB, G0_AFW, G0_AFB, G0_BF,
  // gat1 (H=4, din=512)
  G1_WHW, G1_WHB, G1_W1W, G1_W1B, G1_W2W, G1_W2B, G1_AHW, G1_AHB,
  G1_WFW, G1_WFB, G1_AFW, G1_AFB, G1_BF
};

extern "C" void kernel_launch(void* const* d_in, const int* in_sizes, int n_in,
                              void* d_out, int out_size, void* d_ws, size_t ws_size,
                              hipStream_t stream) {
  (void)in_sizes; (void)n_in; (void)out_size; (void)ws_size;
  auto F = [&](int i) { return (const float*)d_in[i]; };

  // ---- workspace carve (~94 MB total) ----
  size_t off = 0;
  auto alloc = [&](size_t bytes) -> void* {
    void* p = (char*)d_ws + off;
    off += (bytes + 255) & ~(size_t)255;
    return p;
  };
  _Float16* ef16    = (_Float16*)alloc((size_t)NEDGE * 128 * 2);   // 81.92 MB
  float*    xcat32  = (float*)   alloc((size_t)BN * 128 * 4);
  _Float16* xcat16  = (_Float16*)alloc((size_t)BN * 128 * 2);
  float*    nf32    = (float*)   alloc((size_t)BN * 128 * 4);
  _Float16* nf16    = (_Float16*)alloc((size_t)BN * 128 * 2);
  _Float16* nembWp  = (_Float16*)alloc(128 * 128 * 2);   // packed B operands
  _Float16* eembWp  = (_Float16*)alloc(128 * 128 * 2);
  _Float16* g0whWp  = (_Float16*)alloc(8 * 128 * 64 * 2);
  _Float16* g0wfWp  = (_Float16*)alloc(8 * 128 * 64 * 2);
  _Float16* g1whWp  = (_Float16*)alloc(4 * 512 * 64 * 2);
  _Float16* g1wfWp  = (_Float16*)alloc(4 * 128 * 64 * 2);
  float* S1e = (float*)alloc(64 * 4);
  float* S2e = (float*)alloc(64 * 4);
  float* Src = (float*)alloc(64 * 4);
  float* Sdi = (float*)alloc(64 * 4);
  float* Spw = (float*)alloc(128 * 4);
  float* v1_0 = (float*)alloc(8 * 128 * 4);  float* v2_0 = (float*)alloc(8 * 128 * 4);
  float* c1_0 = (float*)alloc(8 * 4);        float* c2_0 = (float*)alloc(8 * 4);
  float* v1_1 = (float*)alloc(4 * 512 * 4);  float* v2_1 = (float*)alloc(4 * 512 * 4);
  float* c1_1 = (float*)alloc(4 * 4);        float* c2_1 = (float*)alloc(4 * 4);
  float* s1_0 = (float*)alloc(8 * BN * 4);   float* s2_0 = (float*)alloc(8 * BN * 4);
  float* s1_1 = (float*)alloc(4 * BN * 4);   float* s2_1 = (float*)alloc(4 * BN * 4);
  float* H0   = (float*)alloc((size_t)8 * BN * 64 * 4);
  float* HP0  = (float*)alloc((size_t)8 * BN * 64 * 4);
  float* X0   = (float*)alloc((size_t)BN * 512 * 4);
  _Float16* X016 = (_Float16*)alloc((size_t)BN * 512 * 2);
  float* H1   = (float*)alloc((size_t)4 * BN * 64 * 4);
  float* HP1  = (float*)alloc((size_t)4 * BN * 64 * 4);
  float* X1   = (float*)alloc((size_t)BN * 64 * 4);

  // ---- pack B operands (f32 -> f16 fragment-contiguous) ----
  auto pack = [&](const float* in, _Float16* out, int K, int N, int Z) {
    long long perZ = (long long)K * N;
    long long total = perZ * Z;
    k_pack_b<<<dim3((unsigned)((total + 255) / 256)), dim3(256), 0, stream>>>(in, out, K, N, total, perZ);
  };
  pack(F(I_NEMBW), nembWp, 128, 128, 1);
  pack(F(I_EEMBW), eembWp, 128, 128, 1);
  pack(F(G0_WHW), g0whWp, 128, 64, 8);
  pack(F(G0_WFW), g0wfWp, 128, 64, 8);
  pack(F(G1_WHW), g1whWp, 512, 64, 4);
  pack(F(G1_WFW), g1wfWp, 128, 64, 4);

  // ---- colsums (scalar-tile collapse) ----
  k_colsum10<<<dim3(1), dim3(256), 0, stream>>>(F(I_EW1W), S1e, 64);
  k_colsum10<<<dim3(1), dim3(256), 0, stream>>>(F(I_EW2W), S2e, 64);
  k_colsum10<<<dim3(1), dim3(256), 0, stream>>>(F(I_RCWW), Src, 64);
  k_colsum10<<<dim3(1), dim3(256), 0, stream>>>(F(I_DIWW), Sdi, 64);
  k_colsum10<<<dim3(1), dim3(256), 0, stream>>>(F(I_PWW),  Spw, 128);

  // ---- node features + node_emb GEMM ----
  k_node_features<<<dim3(BN), dim3(128), 0, stream>>>(
      F(I_XLOC), F(I_XTW), F(I_XD), F(I_XDUAL),
      F(I_NW1W), F(I_NW1B), F(I_NW2W), F(I_NW2B),
      F(I_NW3W), F(I_NW3B), F(I_NW4W), F(I_NW4B), xcat32, xcat16);
  k_gemm<<<dim3(BN / 16, 2, 1), dim3(32), 0, stream>>>(
      xcat16, nembWp, F(I_NEMBB), nf32, nf16, BN, 128, 128, /*relu*/1, 0, 0, 0);

  // ---- fused edge features + edge_emb (f16 output, fits in L2) ----
  k_edge_emb<<<dim3(NEDGE / 16, 2), dim3(32), 0, stream>>>(
      F(I_DIS), F(I_ADJ), S1e, F(I_EW1B), S2e, F(I_EW2B), eembWp, F(I_EEMBB), ef16);

  // ================= GAT layer 0 (H=8, din=128) =================
  k_sprep<<<dim3(8), dim3(128), 0, stream>>>(
      F(G0_W1W), F(G0_W1B), F(G0_W2W), F(G0_W2B), F(G0_AHW), v1_0, v2_0, c1_0, c2_0, 128);
  k_scompute<<<dim3(BN, 8), dim3(32), 0, stream>>>(nf32, v1_0, v2_0, c1_0, c2_0, s1_0, s2_0, 128);
  k_gemm<<<dim3(BN / 16, 1, 8), dim3(32), 0, stream>>>(
      nf16, g0whWp, F(G0_WHB), H0, nullptr, BN, 128, 64, /*none*/0,
      (long long)128 * 64, 64, (long long)BN * 64);
  k_attention<<<dim3(NN, BSZ, 8), dim3(32), 0, stream>>>(
      ef16, g0wfWp, F(G0_WFB), F(G0_BF), F(G0_AFW), F(G0_AFB), F(G0_AHB),
      s1_0, s2_0, H0, F(I_ADJ), HP0);
  k_norm<<<dim3(NN, 8), dim3(256), 0, stream>>>(HP0, X0, 8, /*concat+leaky*/0);
  k_f32_to_f16<<<dim3((BN * 512 + 255) / 256), dim3(256), 0, stream>>>(X0, X016, (long long)BN * 512);

  // ================= GAT layer 1 (H=4, din=512) =================
  k_sprep<<<dim3(4), dim3(512), 0, stream>>>(
      F(G1_W1W), F(G1_W1B), F(G1_W2W), F(G1_W2B), F(G1_AHW), v1_1, v2_1, c1_1, c2_1, 512);
  k_scompute<<<dim3(BN, 4), dim3(32), 0, stream>>>(X0, v1_1, v2_1, c1_1, c2_1, s1_1, s2_1, 512);
  k_gemm<<<dim3(BN / 16, 1, 4), dim3(32), 0, stream>>>(
      X016, g1whWp, F(G1_WHB), H1, nullptr, BN, 512, 64, /*none*/0,
      (long long)512 * 64, 64, (long long)BN * 64);
  k_attention<<<dim3(NN, BSZ, 4), dim3(32), 0, stream>>>(
      ef16, g1wfWp, F(G1_WFB), F(G1_BF), F(G1_AFW), F(G1_AFB), F(G1_AHB),
      s1_1, s2_1, H1, F(I_ADJ), HP1);
  k_norm<<<dim3(NN, 4), dim3(256), 0, stream>>>(HP1, nullptr, 4, /*in-place*/1);
  k_combine1<<<dim3(BN), dim3(64), 0, stream>>>(HP1, X1);

  // ================= tail MLP =================
  k_tail<<<dim3(BSZ), dim3(256), 0, stream>>>(
      X1, F(I_XW1W), F(I_XW1B), F(I_XW2W), F(I_XW2B),
      Src, F(I_RCWB), Sdi, F(I_DIWB),
      F(I_PEW), F(I_PEB), Spw, F(I_PWB),
      F(I_COLW), F(I_COLB), F(I_O1W), F(I_O1B), F(I_O2W), F(I_O2B),
      F(I_RC), F(I_DIST), F(I_PONE), (float*)d_out);
}
